// SceneSAGE_13116830122416
// MI455X (gfx1250) — matrix operand order repaired and checked by reading, not compiled, not run on hardware
//
#include <hip/hip_runtime.h>

// ---------------------------------------------------------------------------
// SceneSAGE on MI455X (gfx1250, wave32, WMMA).
//
// Roofline: dense GEMM work is ~6.5 GFLOP (negligible at WMMA rates); the
// aggregation moves ~330 MB/layer of gathers + 82M f32 atomics, and the whole
// feature matrix (20 MB) lives in the 192 MB L2, so the kernel is L2-bound on
// the scatter/gather. GEMMs therefore run as split-f16 WMMA (Markidis): each
// f32 operand = hi(f16) + lo(f16), D += Ah*Bh + Ah*Bl + Al*Bh with f32
// accumulate -> ~2^-22 relative error (fp32-exact for practical purposes)
// while using only v_wmma_f32_16x16x32_f16. The 3x WMMA cost is free.
// ---------------------------------------------------------------------------

typedef __attribute__((ext_vector_type(16))) _Float16 v16h;
typedef __attribute__((ext_vector_type(8)))  _Float16 v8h;
typedef __attribute__((ext_vector_type(8)))  float    v8f;

#define DH 128   // hidden width (also D_IN)

// ------------------------------ utility kernels ----------------------------

__global__ __launch_bounds__(256) void zero_kernel(float* __restrict__ p, int n) {
    int i = blockIdx.x * 256 + threadIdx.x;
    if (i < n) p[i] = 0.0f;
}

// W [K x C] f32 row-major -> transposed split-f16 pair Wt_hi/Wt_lo [C x K],
// so a B-fragment's 16 consecutive K values are one contiguous 32-byte load.
__global__ __launch_bounds__(256)
void convert_wt_kernel(const float* __restrict__ W,
                       _Float16* __restrict__ Wth, _Float16* __restrict__ Wtl,
                       int K, int C) {
    int i = blockIdx.x * 256 + threadIdx.x;
    if (i >= K * C) return;
    int k = i / C, c = i % C;
    float w = W[i];
    _Float16 hi = (_Float16)w;
    Wth[(size_t)c * K + k] = hi;
    Wtl[(size_t)c * K + k] = (_Float16)(w - (float)hi);
}

__global__ __launch_bounds__(256)
void degree_kernel(const int* __restrict__ dst, float* __restrict__ cnt, int E) {
    int e = blockIdx.x * 256 + threadIdx.x;
    if (e < E) unsafeAtomicAdd(&cnt[dst[e]], 1.0f);
}

// One wave per edge: 32 lanes x float4 = 128 features. Gathers hit L2
// (feature matrix is L2-resident); scatter uses global_atomic_add_f32.
__global__ __launch_bounds__(256)
void scatter_kernel(const float* __restrict__ x, const int* __restrict__ src,
                    const int* __restrict__ dst, float* __restrict__ agg, int E) {
    int e = blockIdx.x * 8 + (threadIdx.x >> 5);
    if (e >= E) return;
    int lane = threadIdx.x & 31;
    int s = src[e], d = dst[e];
    float4 v = *(const float4*)(x + (size_t)s * DH + lane * 4);
    float* p = agg + (size_t)d * DH + lane * 4;
    unsafeAtomicAdd(p + 0, v.x);
    unsafeAtomicAdd(p + 1, v.y);
    unsafeAtomicAdd(p + 2, v.z);
    unsafeAtomicAdd(p + 3, v.w);
}

// ------------------------- fused SAGE matmul (WMMA) ------------------------
// out[row, :] = (agg[row,:]/max(cnt,1)) @ Wl + bias + x[row,:] @ Wr
// Block = 64 rows, 8 waves; each wave owns 16x16 D tiles, K=128 in four
// 16x16x32 steps; split-f16: 6 WMMAs per k-step (3 per product).
template<int NCOL>
__global__ __launch_bounds__(256)
void sage_matmul_kernel(const float* __restrict__ xin,
                        const float* __restrict__ agg,
                        const float* __restrict__ cnt,
                        const _Float16* __restrict__ WlTh,  // [NCOL x 128] f16 hi
                        const _Float16* __restrict__ WlTl,  // [NCOL x 128] f16 lo
                        const _Float16* __restrict__ WrTh,
                        const _Float16* __restrict__ WrTl,
                        const float* __restrict__ bias,     // [NCOL]
                        float* __restrict__ out, int nrows) {
    __shared__ _Float16 sAmh[64 * DH];   // mean hi
    __shared__ _Float16 sAml[64 * DH];   // mean lo
    __shared__ _Float16 sAxh[64 * DH];   // root hi
    __shared__ _Float16 sAxl[64 * DH];   // root lo

    const int tid = threadIdx.x;
    const int rowBase = blockIdx.x * 64;

    // Stage 64x128 of mean (= agg/deg) and x into LDS as split f16.
    for (int i = tid; i < 64 * 32; i += 256) {
        int m  = i >> 5;
        int k4 = (i & 31) << 2;
        int row = rowBase + m;
        int o = m * DH + k4;
        if (row < nrows) {
            float inv = 1.0f / fmaxf(cnt[row], 1.0f);
            float4 av = *(const float4*)(agg + (size_t)row * DH + k4);
            float4 xv = *(const float4*)(xin + (size_t)row * DH + k4);
            float a0 = av.x * inv, a1 = av.y * inv, a2 = av.z * inv, a3 = av.w * inv;
            _Float16 h0 = (_Float16)a0, h1 = (_Float16)a1,
                     h2 = (_Float16)a2, h3 = (_Float16)a3;
            sAmh[o+0] = h0; sAmh[o+1] = h1; sAmh[o+2] = h2; sAmh[o+3] = h3;
            sAml[o+0] = (_Float16)(a0 - (float)h0);
            sAml[o+1] = (_Float16)(a1 - (float)h1);
            sAml[o+2] = (_Float16)(a2 - (float)h2);
            sAml[o+3] = (_Float16)(a3 - (float)h3);
            _Float16 x0 = (_Float16)xv.x, x1 = (_Float16)xv.y,
                     x2 = (_Float16)xv.z, x3 = (_Float16)xv.w;
            sAxh[o+0] = x0; sAxh[o+1] = x1; sAxh[o+2] = x2; sAxh[o+3] = x3;
            sAxl[o+0] = (_Float16)(xv.x - (float)x0);
            sAxl[o+1] = (_Float16)(xv.y - (float)x1);
            sAxl[o+2] = (_Float16)(xv.z - (float)x2);
            sAxl[o+3] = (_Float16)(xv.w - (float)x3);
        } else {
            for (int j = 0; j < 4; ++j) {
                sAmh[o+j] = (_Float16)0.0f; sAml[o+j] = (_Float16)0.0f;
                sAxh[o+j] = (_Float16)0.0f; sAxl[o+j] = (_Float16)0.0f;
            }
        }
    }
    __syncthreads();

    const int w    = tid >> 5;
    const int lane = tid & 31;
    const int h    = lane >> 4;   // half-wave select
    const int mn   = lane & 15;   // A: row M; B/D: column N
    constexpr int NT = NCOL / 16;

    for (int t = w; t < 4 * NT; t += 8) {
        const int rt = t / NT;    // row tile (0..3)
        const int ct = t % NT;    // col tile
        v8f c = {};
#pragma unroll
        for (int ks = 0; ks < 4; ++ks) {
            const int kb = ks * 32;
            const int ao = (rt * 16 + mn) * DH + kb + h * 16;
            // A fragment (16x32 f16): lane holds row mn,
            // K = [kb+16h, kb+16h+16) contiguous in elems 0..15.
            v8h l0, l1;
            l0 = *(const v8h*)&sAmh[ao]; l1 = *(const v8h*)&sAmh[ao + 8];
            v16h amh = __builtin_shufflevector(l0, l1,
                0,1,2,3,4,5,6,7,8,9,10,11,12,13,14,15);
            l0 = *(const v8h*)&sAml[ao]; l1 = *(const v8h*)&sAml[ao + 8];
            v16h aml = __builtin_shufflevector(l0, l1,
                0,1,2,3,4,5,6,7,8,9,10,11,12,13,14,15);
            l0 = *(const v8h*)&sAxh[ao]; l1 = *(const v8h*)&sAxh[ao + 8];
            v16h axh = __builtin_shufflevector(l0, l1,
                0,1,2,3,4,5,6,7,8,9,10,11,12,13,14,15);
            l0 = *(const v8h*)&sAxl[ao]; l1 = *(const v8h*)&sAxl[ao + 8];
            v16h axl = __builtin_shufflevector(l0, l1,
                0,1,2,3,4,5,6,7,8,9,10,11,12,13,14,15);
            // B fragments (32x16): lane holds column mn, 16 contiguous K.
            const size_t bo = (size_t)(ct * 16 + mn) * DH + kb + h * 16;
            v16h blh = *(const v16h*)(WlTh + bo);
            v16h bll = *(const v16h*)(WlTl + bo);
            v16h brh = *(const v16h*)(WrTh + bo);
            v16h brl = *(const v16h*)(WrTl + bo);
            // mean @ Wl : hi*hi + hi*lo + lo*hi
            c = __builtin_amdgcn_wmma_f32_16x16x32_f16(false, amh, false, blh, (short)0, c, false, false);
            c = __builtin_amdgcn_wmma_f32_16x16x32_f16(false, amh, false, bll, (short)0, c, false, false);
            c = __builtin_amdgcn_wmma_f32_16x16x32_f16(false, aml, false, blh, (short)0, c, false, false);
            // x @ Wr : hi*hi + hi*lo + lo*hi
            c = __builtin_amdgcn_wmma_f32_16x16x32_f16(false, axh, false, brh, (short)0, c, false, false);
            c = __builtin_amdgcn_wmma_f32_16x16x32_f16(false, axh, false, brl, (short)0, c, false, false);
            c = __builtin_amdgcn_wmma_f32_16x16x32_f16(false, axl, false, brh, (short)0, c, false, false);
        }
        // D layout: lane holds column mn, rows 8h+r in accumulator r.
        const float bv = bias[ct * 16 + mn];
        float* po = out + (size_t)(rowBase + rt * 16 + h * 8) * NCOL + ct * 16 + mn;
        if (rowBase + 64 <= nrows) {            // full tile: straight-line stores
#pragma unroll
            for (int r = 0; r < 8; ++r)
                po[(size_t)r * NCOL] = c[r] + bv;
        } else {
#pragma unroll
            for (int r = 0; r < 8; ++r) {
                const int row = rowBase + rt * 16 + h * 8 + r;
                if (row < nrows) po[(size_t)r * NCOL] = c[r] + bv;
            }
        }
    }
}

// -------------------------- ReLU + LayerNorm (in place) --------------------
// One wave32 per row of 128 features; 4 features per lane; shuffle reduce.
__global__ __launch_bounds__(256)
void relu_ln_kernel(float* __restrict__ hbuf, const float* __restrict__ g,
                    const float* __restrict__ b, int nrows) {
    const int row = blockIdx.x * 8 + (threadIdx.x >> 5);
    if (row >= nrows) return;
    const int lane = threadIdx.x & 31;
    float4 v = *(const float4*)(hbuf + (size_t)row * DH + lane * 4);
    v.x = fmaxf(v.x, 0.0f); v.y = fmaxf(v.y, 0.0f);
    v.z = fmaxf(v.z, 0.0f); v.w = fmaxf(v.w, 0.0f);
    float s  = v.x + v.y + v.z + v.w;
    float s2 = v.x * v.x + v.y * v.y + v.z * v.z + v.w * v.w;
#pragma unroll
    for (int off = 16; off > 0; off >>= 1) {
        s  += __shfl_xor(s,  off, 32);
        s2 += __shfl_xor(s2, off, 32);
    }
    const float mu  = s  * (1.0f / 128.0f);
    float       var = s2 * (1.0f / 128.0f) - mu * mu;
    var = fmaxf(var, 0.0f);
    const float rs = rsqrtf(var + 1e-5f);
    const float4 gg = *(const float4*)(g + lane * 4);
    const float4 bb = *(const float4*)(b + lane * 4);
    float4 o;
    o.x = (v.x - mu) * rs * gg.x + bb.x;
    o.y = (v.y - mu) * rs * gg.y + bb.y;
    o.z = (v.z - mu) * rs * gg.z + bb.z;
    o.w = (v.w - mu) * rs * gg.w + bb.w;
    *(float4*)(hbuf + (size_t)row * DH + lane * 4) = o;
}

// ------------------------------- launcher ----------------------------------

extern "C" void kernel_launch(void* const* d_in, const int* in_sizes, int n_in,
                              void* d_out, int out_size, void* d_ws, size_t ws_size,
                              hipStream_t stream) {
    const float* x   = (const float*)d_in[0];
    const int*   ei  = (const int*)  d_in[1];
    const float* Wl0 = (const float*)d_in[2];
    const float* bl0 = (const float*)d_in[3];
    const float* Wr0 = (const float*)d_in[4];
    const float* Wl1 = (const float*)d_in[5];
    const float* bl1 = (const float*)d_in[6];
    const float* Wr1 = (const float*)d_in[7];
    const float* Wl2 = (const float*)d_in[8];
    const float* bl2 = (const float*)d_in[9];
    const float* Wr2 = (const float*)d_in[10];
    const float* g0  = (const float*)d_in[11];
    const float* b0  = (const float*)d_in[12];
    const float* g1  = (const float*)d_in[13];
    const float* b1  = (const float*)d_in[14];

    const int N = in_sizes[0] / DH;   // 40000
    const int E = in_sizes[1] / 2;    // 640000
    const int* src = ei;
    const int* dst = ei + E;
    const int D_OUT = 64;

    // Workspace carve-out (256B aligned): agg, cnt, two ping-pong feature
    // buffers, split-f16 transposed weights. Total ~62 MB.
    char* ws = (char*)d_ws;
    size_t off = 0;
    auto carve = [&](size_t bytes) -> void* {
        void* p = ws + off;
        off += (bytes + 255) & ~(size_t)255;
        return p;
    };
    float*    agg   = (float*)   carve((size_t)N * DH * 4);
    float*    cnt   = (float*)   carve((size_t)N * 4);
    float*    P0    = (float*)   carve((size_t)N * DH * 4);
    float*    P1    = (float*)   carve((size_t)N * DH * 4);
    _Float16* WlT0h = (_Float16*)carve((size_t)DH * DH * 2);
    _Float16* WlT0l = (_Float16*)carve((size_t)DH * DH * 2);
    _Float16* WrT0h = (_Float16*)carve((size_t)DH * DH * 2);
    _Float16* WrT0l = (_Float16*)carve((size_t)DH * DH * 2);
    _Float16* WlT1h = (_Float16*)carve((size_t)DH * DH * 2);
    _Float16* WlT1l = (_Float16*)carve((size_t)DH * DH * 2);
    _Float16* WrT1h = (_Float16*)carve((size_t)DH * DH * 2);
    _Float16* WrT1l = (_Float16*)carve((size_t)DH * DH * 2);
    _Float16* WlT2h = (_Float16*)carve((size_t)DH * D_OUT * 2);
    _Float16* WlT2l = (_Float16*)carve((size_t)DH * D_OUT * 2);
    _Float16* WrT2h = (_Float16*)carve((size_t)DH * D_OUT * 2);
    _Float16* WrT2l = (_Float16*)carve((size_t)DH * D_OUT * 2);
    (void)ws_size; (void)n_in; (void)out_size;

    const int featN   = N * DH;
    const int blkFeat = (featN + 255) / 256;
    const int blkRow  = (N + 63) / 64;
    const int blkLN   = (N + 7) / 8;
    const int blkEdge = (E + 7) / 8;

    // ---- one-time per call: degree + weight convert/transpose ----
    zero_kernel<<<(N + 255) / 256, 256, 0, stream>>>(cnt, N);
    degree_kernel<<<(E + 255) / 256, 256, 0, stream>>>(dst, cnt, E);
    convert_wt_kernel<<<(DH * DH + 255) / 256, 256, 0, stream>>>(Wl0, WlT0h, WlT0l, DH, DH);
    convert_wt_kernel<<<(DH * DH + 255) / 256, 256, 0, stream>>>(Wr0, WrT0h, WrT0l, DH, DH);
    convert_wt_kernel<<<(DH * DH + 255) / 256, 256, 0, stream>>>(Wl1, WlT1h, WlT1l, DH, DH);
    convert_wt_kernel<<<(DH * DH + 255) / 256, 256, 0, stream>>>(Wr1, WrT1h, WrT1l, DH, DH);
    convert_wt_kernel<<<(DH * D_OUT + 255) / 256, 256, 0, stream>>>(Wl2, WlT2h, WlT2l, DH, D_OUT);
    convert_wt_kernel<<<(DH * D_OUT + 255) / 256, 256, 0, stream>>>(Wr2, WrT2h, WrT2l, DH, D_OUT);

    // ---- layer 0 ----
    zero_kernel<<<blkFeat, 256, 0, stream>>>(agg, featN);
    scatter_kernel<<<blkEdge, 256, 0, stream>>>(x, src, dst, agg, E);
    sage_matmul_kernel<128><<<blkRow, 256, 0, stream>>>(
        x, agg, cnt, WlT0h, WlT0l, WrT0h, WrT0l, bl0, P0, N);
    relu_ln_kernel<<<blkLN, 256, 0, stream>>>(P0, g0, b0, N);

    // ---- layer 1 ----
    zero_kernel<<<blkFeat, 256, 0, stream>>>(agg, featN);
    scatter_kernel<<<blkEdge, 256, 0, stream>>>(P0, src, dst, agg, E);
    sage_matmul_kernel<128><<<blkRow, 256, 0, stream>>>(
        P0, agg, cnt, WlT1h, WlT1l, WrT1h, WrT1l, bl1, P1, N);
    relu_ln_kernel<<<blkLN, 256, 0, stream>>>(P1, g1, b1, N);

    // ---- layer 2 (no LN, direct to output) ----
    zero_kernel<<<blkFeat, 256, 0, stream>>>(agg, featN);
    scatter_kernel<<<blkEdge, 256, 0, stream>>>(P1, src, dst, agg, E);
    sage_matmul_kernel<64><<<blkRow, 256, 0, stream>>>(
        P1, agg, cnt, WlT2h, WlT2l, WrT2h, WrT2l, bl2, (float*)d_out, N);
}
